// AttentionWavelet_8031588843680
// MI455X (gfx1250) — compile-verified, matrix operand order checked by reading, not run on hardware
//
#include <hip/hip_runtime.h>
#include <hip/hip_fp16.h>
#include <math.h>

typedef __attribute__((ext_vector_type(16))) _Float16 v16h;
typedef __attribute__((ext_vector_type(8)))  float    v8f;

// bior4.4 decomposition filters (length 10)
__constant__ float c_lo[10] = {
    0.0f, 0.03782845550726404f, -0.023849465019556843f,
    -0.11062440441843718f, 0.37740285561283066f, 0.8526986790088938f,
    0.37740285561283066f, -0.11062440441843718f,
    -0.023849465019556843f, 0.03782845550726404f};
__constant__ float c_hi[10] = {
    0.0f, -0.06453888262869706f, 0.04068941760916406f,
    0.41809227322161724f, -0.7884856164055829f, 0.41809227322161724f,
    0.04068941760916406f, -0.06453888262869706f, 0.0f, 0.0f};

// ---------------- branch-free tanh (CDNA5 v_tanh when available) ----------
__device__ __forceinline__ float tanh_fast(float x) {
#if __has_builtin(__builtin_amdgcn_tanhf)
  return __builtin_amdgcn_tanhf(x);
#else
  float ax = __builtin_fabsf(x);
  float e2 = __expf(-2.0f * ax);
  float r  = (1.0f - e2) * __builtin_amdgcn_rcpf(1.0f + e2);
  return copysignf(r, x);
#endif
}

__device__ __forceinline__ _Float16 tanh_h(float x) {
#if __has_builtin(__builtin_amdgcn_tanh_f16)
  return __builtin_amdgcn_tanh_f16((_Float16)x);
#else
  return (_Float16)tanh_fast(x);
#endif
}

// ---------------- DWT: row pass (along W), periodized ----------------
// x: (P, H, W) with P = B*C planes; lo/hi: (P, H, W/2)
__global__ void dwt_row_kernel(const float* __restrict__ x,
                               float* __restrict__ lo, float* __restrict__ hi,
                               int W, long total) {
  long idx = (long)blockIdx.x * blockDim.x + threadIdx.x;
  if (idx >= total) return;
  int n2 = W >> 1;
  int n = (int)(idx % n2);
  long rh = idx / n2;                 // plane*H + h (rows contiguous)
  const float* row = x + rh * (long)W;
  float aL = 0.f, aH = 0.f;
  int base = 2 * n - 5;
#pragma unroll
  for (int l = 0; l < 10; ++l) {
    int k = base + l;
    if (k < 0) k += W;
    if (k >= W) k -= W;
    float v = row[k];
    aL = fmaf(v, c_lo[l], aL);
    aH = fmaf(v, c_hi[l], aH);
  }
  lo[rh * n2 + n] = aL;
  hi[rh * n2 + n] = aH;
}

// ---------------- DWT: column pass (along H), periodized ----------------
// lo/hi: (P, H, W2)  ->  ll: (P, H/2, W2), bands: (P, 3, H/2, W2) [lh,hl,hh]
__global__ void dwt_col_kernel(const float* __restrict__ lo,
                               const float* __restrict__ hi,
                               float* __restrict__ ll, float* __restrict__ bands,
                               int H, int W2, long total) {
  long idx = (long)blockIdx.x * blockDim.x + threadIdx.x;
  if (idx >= total) return;
  int H2 = H >> 1;
  int w = (int)(idx % W2);
  long t = idx / W2;
  int h2 = (int)(t % H2);
  long plane = t / H2;
  const float* pl = lo + plane * (long)H * W2 + w;
  const float* ph = hi + plane * (long)H * W2 + w;
  float a_ll = 0.f, a_lh = 0.f, a_hl = 0.f, a_hh = 0.f;
  int base = 2 * h2 - 5;
#pragma unroll
  for (int l = 0; l < 10; ++l) {
    int k = base + l;
    if (k < 0) k += H;
    if (k >= H) k -= H;
    float vl = pl[(long)k * W2];
    float vh = ph[(long)k * W2];
    a_ll = fmaf(vl, c_lo[l], a_ll);
    a_lh = fmaf(vl, c_hi[l], a_lh);
    a_hl = fmaf(vh, c_lo[l], a_hl);
    a_hh = fmaf(vh, c_hi[l], a_hh);
  }
  long hw = (long)h2 * W2 + w;
  long H2W2 = (long)H2 * W2;
  ll[plane * H2W2 + hw] = a_ll;
  float* bp = bands + plane * 3 * H2W2 + hw;
  bp[0]        = a_lh;    // band 0 = LH
  bp[H2W2]     = a_hl;    // band 1 = HL
  bp[2 * H2W2] = a_hh;    // band 2 = HH
}

// ---------------- attention fuse: ll*(1+softmax_c(LH+HL)) ----------------
__global__ void att_fuse_kernel(const float* __restrict__ ll3,
                                const float* __restrict__ bands3,
                                float* __restrict__ llf,
                                int HW, long total) {   // total = B*HW, C==3
  long idx = (long)blockIdx.x * blockDim.x + threadIdx.x;
  if (idx >= total) return;
  int b = (int)(idx / HW);
  int r = (int)(idx % HW);
  float e[3];
#pragma unroll
  for (int c = 0; c < 3; ++c) {
    long pb = ((long)(b * 3 + c)) * 3 * HW + r;   // (B,C,3,HW)
    e[c] = bands3[pb] + bands3[pb + HW];          // LH + HL
  }
  float m = fmaxf(e[0], fmaxf(e[1], e[2]));
  float x0 = __expf(e[0] - m), x1 = __expf(e[1] - m), x2 = __expf(e[2] - m);
  float inv = 1.0f / (x0 + x1 + x2);
  float a[3] = {x0 * inv, x1 * inv, x2 * inv};
#pragma unroll
  for (int c = 0; c < 3; ++c) {
    long pa = ((long)(b * 3 + c)) * HW + r;
    float v = ll3[pa];
    llf[pa] = v + v * a[c];
  }
}

// ---------------- per-channel 1->32->32->32->1 MLP via WMMA ----------------
// Hidden ordering between layers is internal, so we fold the D->B layout
// permutation pi into the layer-3 / layer-4 weight loads:
//   pi(slot j of this lane) = the D-matrix row this lane's slot j receives
//   = (j<8) ? half*8+j : 16+half*8+(j-8)
// With that, D_lo/D_hi feed the next B operand directly in-register:
// no cross-lane exchange between layers. Biases ride in as the WMMA C operand.
__device__ __forceinline__ unsigned pack2h(_Float16 a, _Float16 b) {
  union { _Float16 h[2]; unsigned u; } t;
  t.h[0] = a; t.h[1] = b;
  return t.u;
}

__global__ __launch_bounds__(256)
void mlp_wmma_kernel(const float* __restrict__ xin, float* __restrict__ out,
                     const float* __restrict__ w1, const float* __restrict__ b1,
                     const float* __restrict__ w2, const float* __restrict__ b2,
                     const float* __restrict__ w3, const float* __restrict__ b3,
                     const float* __restrict__ w4, const float* __restrict__ b4,
                     int C, int hwShift, int tiles) {
  const int lane = threadIdx.x & 31;
  const int wid  = threadIdx.x >> 5;
  const int c    = blockIdx.x;
  const int half = lane >> 4;       // which 16-lane half of the wave32
  const int p16  = lane & 15;       // pixel slot / matrix row within half
  const int k0   = half * 16;       // this lane's K-range start for A/B operands
  const int HWm1 = (1 << hwShift) - 1;

  union VH { v16h v; _Float16 h[16]; unsigned u[8]; };

  // ---- wave-resident weights (loaded once per wave) ----
  VH A2lo, A2hi, A3lo, A3hi;        // A operand = W[o, k], f16
  const float* w2c = w2 + (long)c * 32 * 32;
  const float* w3c = w3 + (long)c * 32 * 32;
  float w1v[16], b1v[16], w4v[16];
#pragma unroll
  for (int j = 0; j < 16; ++j) {
    // permutation pi for this lane's slot j (layer-2 D row landing in slot j)
    const int pk = (j < 8) ? (half * 8 + j) : (16 + half * 8 + (j - 8));
    A2lo.h[j] = (_Float16)w2c[p16 * 32 + k0 + j];        // o = 0..15, K natural
    A2hi.h[j] = (_Float16)w2c[(16 + p16) * 32 + k0 + j]; // o = 16..31
    A3lo.h[j] = (_Float16)w3c[p16 * 32 + pk];            // K permuted by pi
    A3hi.h[j] = (_Float16)w3c[(16 + p16) * 32 + pk];
    w1v[j] = w1[c * 32 + k0 + j];                        // natural order
    b1v[j] = b1[c * 32 + k0 + j];
    w4v[j] = w4[c * 32 + pk];                            // permuted by pi
  }
  // biases as WMMA C operand: C[m][n] = b[m]; lane-half h, VGPR j -> m = h*8+j
  v8f c2lo, c2hi, c3lo, c3hi;
#pragma unroll
  for (int j = 0; j < 8; ++j) {
    c2lo[j] = b2[c * 32 + half * 8 + j];
    c2hi[j] = b2[c * 32 + 16 + half * 8 + j];
    c3lo[j] = b3[c * 32 + half * 8 + j];
    c3hi[j] = b3[c * 32 + 16 + half * 8 + j];
  }
  const float b4s = b4[c];

  const int wavesPerC = gridDim.y * (blockDim.x >> 5);
  for (int t = blockIdx.y * (blockDim.x >> 5) + wid; t < tiles; t += wavesPerC) {
    // 16 pixels per tile; both lane-halves load the same pixel (broadcast hit)
    int q = t * 16 + p16;
    int bimg = q >> hwShift;
    int r = q & HWm1;
    long addr = ((long)(bimg * C + c) << hwShift) + r;
    float xv = xin[addr];

    // layer 1: h1[k] = tanh(x*w1[k]+b1[k]); lane holds K=k0..k0+15 of pixel p16
    VH Bm;
#pragma unroll
    for (int j = 0; j < 16; ++j)
      Bm.h[j] = tanh_h(fmaf(xv, w1v[j], b1v[j]));

    // layer 2: D = W2 x h1 + b2  (bias in C operand)
    v8f Dlo = __builtin_amdgcn_wmma_f32_16x16x32_f16(false, A2lo.v, false, Bm.v,
                                                     (short)0, c2lo, false, false);
    v8f Dhi = __builtin_amdgcn_wmma_f32_16x16x32_f16(false, A2hi.v, false, Bm.v,
                                                     (short)0, c2hi, false, false);
    // tanh + pack straight back into the B operand (pi-folded, no shuffle)
#pragma unroll
    for (int j = 0; j < 4; ++j) {
      Bm.u[j]     = pack2h(tanh_h(Dlo[2 * j]), tanh_h(Dlo[2 * j + 1]));
      Bm.u[4 + j] = pack2h(tanh_h(Dhi[2 * j]), tanh_h(Dhi[2 * j + 1]));
    }

    // layer 3 (A3 columns pre-permuted by pi)
    Dlo = __builtin_amdgcn_wmma_f32_16x16x32_f16(false, A3lo.v, false, Bm.v,
                                                 (short)0, c3lo, false, false);
    Dhi = __builtin_amdgcn_wmma_f32_16x16x32_f16(false, A3hi.v, false, Bm.v,
                                                 (short)0, c3hi, false, false);
#pragma unroll
    for (int j = 0; j < 4; ++j) {
      Bm.u[j]     = pack2h(tanh_h(Dlo[2 * j]), tanh_h(Dlo[2 * j + 1]));
      Bm.u[4 + j] = pack2h(tanh_h(Dhi[2 * j]), tanh_h(Dhi[2 * j + 1]));
    }

    // layer 4: y[p] = w4 . h3[:,p] + b4 (w4 pi-permuted; half-sum via 1 shuffle)
    float partial = 0.f;
#pragma unroll
    for (int j = 0; j < 16; ++j)
      partial = fmaf((float)Bm.h[j], w4v[j], partial);
    float other = __shfl_xor(partial, 16, 32);
    float y = partial + other + b4s;
    if (half == 0) out[addr] = y;
  }
}

// ======================================================================
extern "C" void kernel_launch(void* const* d_in, const int* in_sizes, int n_in,
                              void* d_out, int out_size, void* d_ws, size_t ws_size,
                              hipStream_t stream) {
  const float* x     = (const float*)d_in[0];
  const float* yl_w1 = (const float*)d_in[1];
  const float* yl_b1 = (const float*)d_in[2];
  const float* yl_w2 = (const float*)d_in[3];
  const float* yl_b2 = (const float*)d_in[4];
  const float* yl_w3 = (const float*)d_in[5];
  const float* yl_b3 = (const float*)d_in[6];
  const float* yl_w4 = (const float*)d_in[7];
  const float* yl_b4 = (const float*)d_in[8];
  const float* yh_w1 = (const float*)d_in[9];
  const float* yh_b1 = (const float*)d_in[10];
  const float* yh_w2 = (const float*)d_in[11];
  const float* yh_b2 = (const float*)d_in[12];
  const float* yh_w3 = (const float*)d_in[13];
  const float* yh_b3 = (const float*)d_in[14];
  const float* yh_w4 = (const float*)d_in[15];
  const float* yh_b4 = (const float*)d_in[16];
  float* out = (float*)d_out;
  float* ws  = (float*)d_ws;

  const int B = 4, C = 3, P = B * C;

  // workspace layout (floats); lo/hi reused across levels (level-1 largest)
  size_t off = 0;
  float* lo     = ws + off; off += (size_t)P * 512 * 256;
  float* hi     = ws + off; off += (size_t)P * 512 * 256;
  float* ll1    = ws + off; off += (size_t)P * 256 * 256;
  float* bands1 = ws + off; off += (size_t)B * 9 * 256 * 256;
  float* ll2    = ws + off; off += (size_t)P * 128 * 128;
  float* bands2 = ws + off; off += (size_t)B * 9 * 128 * 128;
  float* ll3    = ws + off; off += (size_t)P * 64 * 64;
  float* bands3 = ws + off; off += (size_t)B * 9 * 64 * 64;
  float* llf    = ws + off; off += (size_t)P * 64 * 64;
  (void)ws_size; (void)in_sizes; (void)n_in; (void)out_size;

  const int TB = 256;
  // ---- level 1: 512x512 -> 256x256 ----
  {
    long tr = (long)P * 512 * 256;
    dwt_row_kernel<<<(unsigned)((tr + TB - 1) / TB), TB, 0, stream>>>(x, lo, hi, 512, tr);
    long tc = (long)P * 256 * 256;
    dwt_col_kernel<<<(unsigned)((tc + TB - 1) / TB), TB, 0, stream>>>(lo, hi, ll1, bands1, 512, 256, tc);
  }
  // ---- level 2: 256x256 -> 128x128 ----
  {
    long tr = (long)P * 256 * 128;
    dwt_row_kernel<<<(unsigned)((tr + TB - 1) / TB), TB, 0, stream>>>(ll1, lo, hi, 256, tr);
    long tc = (long)P * 128 * 128;
    dwt_col_kernel<<<(unsigned)((tc + TB - 1) / TB), TB, 0, stream>>>(lo, hi, ll2, bands2, 256, 128, tc);
  }
  // ---- level 3: 128x128 -> 64x64 ----
  {
    long tr = (long)P * 128 * 64;
    dwt_row_kernel<<<(unsigned)((tr + TB - 1) / TB), TB, 0, stream>>>(ll2, lo, hi, 128, tr);
    long tc = (long)P * 64 * 64;
    dwt_col_kernel<<<(unsigned)((tc + TB - 1) / TB), TB, 0, stream>>>(lo, hi, ll3, bands3, 128, 64, tc);
  }
  // ---- attention fuse on LL3 ----
  {
    long ta = (long)B * 64 * 64;
    att_fuse_kernel<<<(unsigned)((ta + TB - 1) / TB), TB, 0, stream>>>(ll3, bands3, llf, 64 * 64, ta);
  }

  // ---- MLPs (WMMA) ----
  // out_xe : (4,3,64,64)
  {
    int hwShift = 12, tiles = (B << hwShift) / 16;  // 1024
    mlp_wmma_kernel<<<dim3(3, 32), 256, 0, stream>>>(
        llf, out, yl_w1, yl_b1, yl_w2, yl_b2, yl_w3, yl_b3, yl_w4, yl_b4,
        3, hwShift, tiles);
  }
  long o = (long)B * 3 * 64 * 64;  // 49152
  // outs[0] : (4,9,256,256)
  {
    int hwShift = 16, tiles = (B << hwShift) / 16;  // 16384
    mlp_wmma_kernel<<<dim3(9, 128), 256, 0, stream>>>(
        bands1, out + o,
        yh_w1 + 0, yh_b1 + 0, yh_w2 + 0, yh_b2 + 0,
        yh_w3 + 0, yh_b3 + 0, yh_w4 + 0, yh_b4 + 0,
        9, hwShift, tiles);
    o += (long)B * 9 * (1 << hwShift);
  }
  // outs[1] : (4,9,128,128)
  {
    int hwShift = 14, tiles = (B << hwShift) / 16;  // 4096
    mlp_wmma_kernel<<<dim3(9, 64), 256, 0, stream>>>(
        bands2, out + o,
        yh_w1 + 9 * 32, yh_b1 + 9 * 32, yh_w2 + 9 * 1024, yh_b2 + 9 * 32,
        yh_w3 + 9 * 1024, yh_b3 + 9 * 32, yh_w4 + 9 * 32, yh_b4 + 9,
        9, hwShift, tiles);
    o += (long)B * 9 * (1 << hwShift);
  }
  // outs[2] : (4,9,64,64)
  {
    int hwShift = 12, tiles = (B << hwShift) / 16;  // 1024
    mlp_wmma_kernel<<<dim3(9, 32), 256, 0, stream>>>(
        bands3, out + o,
        yh_w1 + 2 * 9 * 32, yh_b1 + 2 * 9 * 32, yh_w2 + 2 * 9 * 1024, yh_b2 + 2 * 9 * 32,
        yh_w3 + 2 * 9 * 1024, yh_b3 + 2 * 9 * 32, yh_w4 + 2 * 9 * 32, yh_b4 + 2 * 9,
        9, hwShift, tiles);
  }
}
